// MHCVP_62319975465194
// MI455X (gfx1250) — compile-verified, matrix operand order checked by reading, not run on hardware
//
#include <hip/hip_runtime.h>

#define BATCH 4
#define CDIM  64
#define HDIM  192
#define WDIM  640
#define NDISP 10
#define NHEAD 8
#define HSZ   8
#define OUTC  74
#define LN_EPS 1e-5f

typedef __attribute__((ext_vector_type(16))) _Float16 v16h;
typedef __attribute__((ext_vector_type(8)))  _Float16 v8h;
typedef __attribute__((ext_vector_type(8)))  float    v8f;

// ---------------- pass 1: per-block partial sums (deterministic) ----------------
#define NBLK  240
#define CHUNK 32768   // NBLK * CHUNK == CDIM*HDIM*WDIM == 7,864,320

__global__ __launch_bounds__(256) void stats_partial_kernel(
    const float* __restrict__ img, float* __restrict__ part) {
  const int b = blockIdx.y;
  const int chunk = blockIdx.x;
  const float4* p = (const float4*)(img + (size_t)b * (CDIM * HDIM * WDIM) +
                                    (size_t)chunk * CHUNK);
  float s = 0.f, q = 0.f;
  for (int i = threadIdx.x; i < CHUNK / 4; i += 256) {
    float4 v = p[i];
    s += v.x + v.y + v.z + v.w;
    q += v.x * v.x + v.y * v.y + v.z * v.z + v.w * v.w;
  }
  __shared__ float sh_s[256];
  __shared__ float sh_q[256];
  sh_s[threadIdx.x] = s;
  sh_q[threadIdx.x] = q;
  __syncthreads();
  for (int off = 128; off > 0; off >>= 1) {
    if (threadIdx.x < off) {
      sh_s[threadIdx.x] += sh_s[threadIdx.x + off];
      sh_q[threadIdx.x] += sh_q[threadIdx.x + off];
    }
    __syncthreads();
  }
  if (threadIdx.x == 0) {
    part[(b * NBLK + chunk) * 2 + 0] = sh_s[0];
    part[(b * NBLK + chunk) * 2 + 1] = sh_q[0];
  }
}

// ---------------- pass 2: reduce partials -> mean, 1/(std+eps) ----------------
__global__ __launch_bounds__(256) void stats_reduce_kernel(
    const float* __restrict__ part, float* __restrict__ stats) {
  // blockIdx.x in [0,8): 0..3 = left batches, 4..7 = right batches
  const float* p = part + (size_t)blockIdx.x * (NBLK * 2);
  float s = 0.f, q = 0.f;
  for (int i = threadIdx.x; i < NBLK; i += 256) {
    s += p[i * 2 + 0];
    q += p[i * 2 + 1];
  }
  __shared__ float sh_s[256];
  __shared__ float sh_q[256];
  sh_s[threadIdx.x] = s;
  sh_q[threadIdx.x] = q;
  __syncthreads();
  for (int off = 128; off > 0; off >>= 1) {
    if (threadIdx.x < off) {
      sh_s[threadIdx.x] += sh_s[threadIdx.x + off];
      sh_q[threadIdx.x] += sh_q[threadIdx.x + off];
    }
    __syncthreads();
  }
  if (threadIdx.x == 0) {
    const float N = (float)(CDIM * HDIM * WDIM);
    float sum = sh_s[0], sq = sh_q[0];
    float mean = sum / N;
    float var = (sq - sum * sum / N) / (N - 1.0f);
    var = var > 0.f ? var : 0.f;
    float inv = 1.0f / (sqrtf(var) + LN_EPS);
    stats[blockIdx.x * 2 + 0] = mean;
    stats[blockIdx.x * 2 + 1] = inv;
  }
}

// ---------------- pass 3: fused LN + posenc + WMMA banded-correlation ----------------
#define TILE  128
#define HALO  16
#define RCOLS (TILE + HALO)   // 144
#define RSTR  72              // 64 ch + 8 pad halves -> 144B row, 16B-aligned head groups
#define LSTR  72
#define MSTR  17              // padded C-tile row to dodge LDS bank conflicts

__global__ __launch_bounds__(256) void fused_kernel(
    const float* __restrict__ left, const float* __restrict__ right,
    const float* __restrict__ homo,
    const float* __restrict__ lnw, const float* __restrict__ lnb,
    const float* __restrict__ gpw, const float* __restrict__ gpb,
    const float* __restrict__ stats, float* __restrict__ out) {
  __shared__ __align__(16) _Float16 rbuf[RCOLS * RSTR];
  __shared__ __align__(16) _Float16 lbuf[TILE * LSTR];
  __shared__ __align__(16) _Float16 zrow[CDIM];   // zero source for lanes 16..31
  __shared__ float mbuf[8 * 2 * 16 * MSTR];
  __shared__ float invfS[16];
  __shared__ float gpwS[NDISP * NHEAD];
  __shared__ float gpbS[NDISP];
  __shared__ float lnwS[CDIM];
  __shared__ float lnbS[CDIM];

  const int tid = threadIdx.x;
  const int b  = blockIdx.z;
  const int h  = blockIdx.y;
  const int w0 = blockIdx.x * TILE;

  if (tid < 16) invfS[tid] = exp2f(-0.47774103f * (float)tid);  // 200^(-p/16)
  if (tid < NDISP * NHEAD) gpwS[tid] = gpw[tid];
  if (tid < NDISP) gpbS[tid] = gpb[tid];
  if (tid < CDIM) {
    lnwS[tid] = lnw[tid];
    lnbS[tid] = lnb[tid];
    zrow[tid] = (_Float16)0.0f;
  }
  __syncthreads();

  const float meanL = stats[(0 * BATCH + b) * 2 + 0];
  const float invL  = stats[(0 * BATCH + b) * 2 + 1];
  const float meanR = stats[(1 * BATCH + b) * 2 + 0];
  const float invR  = stats[(1 * BATCH + b) * 2 + 1];
  const float h00 = homo[b * 9 + 0], h01 = homo[b * 9 + 1], h02 = homo[b * 9 + 2];
  const float h10 = homo[b * 9 + 3], h11 = homo[b * 9 + 4], h12 = homo[b * 9 + 5];
  const float pxh = (float)(h - HDIM / 2);
  const float rxh = h00 * pxh + h02;   // homography row-0 partial (y term added per col)
  const float ryh = h10 * pxh + h12;

  const size_t planeHW = (size_t)HDIM * WDIM;
  const float* Lb = left  + (size_t)b * CDIM * planeHW + (size_t)h * WDIM;
  const float* Rb = right + (size_t)b * CDIM * planeHW + (size_t)h * WDIM;
  float*       Ob = out   + (size_t)b * OUTC * planeHW + (size_t)h * WDIM;

  __builtin_prefetch(&Lb[w0], 0, 0);
  __builtin_prefetch(&Rb[w0], 0, 0);

  // ---- stage right: LN + warped posenc, halo with wraparound (jnp.roll) ----
  // float4 groups: halo base w0-16 is 16-aligned; the only wrap (w0==0) maps a
  // 16-aligned contiguous region to W-16..W-1, so groups stay contiguous.
  for (int idx = tid; idx < CDIM * (RCOLS / 4); idx += 256) {
    int ch = idx / (RCOLS / 4);
    int g4 = idx - ch * (RCOLS / 4);
    int col = g4 * 4;
    int wb = w0 - HALO + col;
    if (wb < 0) wb += WDIM;
    float4 v = *(const float4*)&Rb[(size_t)ch * planeHW + wb];
    int k = ch >> 1, p = ch >> 2;
    float fr = invfS[p];
    float vv[4] = {v.x, v.y, v.z, v.w};
#pragma unroll
    for (int c = 0; c < 4; ++c) {
      float nv = (vv[c] - meanR) * invR * lnwS[ch] + lnbS[ch];
      float pyw = (float)(wb + c - WDIM / 2);
      float arg = ((ch & 1) ? (ryh + h11 * pyw) : (rxh + h01 * pyw)) * fr;
      float pe = (k & 1) ? __cosf(arg) : __sinf(arg);
      rbuf[(col + c) * RSTR + ch] = (_Float16)(nv + pe);
    }
  }

  // ---- stage left: LN + separable posenc; also copy raw left to output ----
  for (int idx = tid; idx < CDIM * (TILE / 4); idx += 256) {
    int ch = idx >> 5;
    int px = (idx & 31) * 4;
    int w = w0 + px;
    float4 v = *(const float4*)&Lb[(size_t)ch * planeHW + w];
    *(float4*)&Ob[(size_t)ch * planeHW + w] = v;  // concat channels 0..63 = raw left
    int k = ch >> 1, p = ch >> 2;
    float fr = invfS[p];
    float vv[4] = {v.x, v.y, v.z, v.w};
#pragma unroll
    for (int c = 0; c < 4; ++c) {
      float nv = (vv[c] - meanL) * invL * lnwS[ch] + lnbS[ch];
      float arg = ((ch & 1) ? (float)(w + c - WDIM / 2) : pxh) * fr;
      float pe = (k & 1) ? __cosf(arg) : __sinf(arg);
      lbuf[(px + c) * LSTR + ch] = (_Float16)(nv + pe);
    }
  }
  __syncthreads();

  // ---- per-wave 16-pixel subtile: per-head banded GEMM via WMMA ----
  const int wave = tid >> 5;
  const int lane = tid & 31;
  const int m = lane & 15;        // pixel row within subtile
  const int dHalf = lane >> 4;    // disparity parity
  const int sBase = wave * 16;    // subtile pixel base within tile
  float acc[5] = {0.f, 0.f, 0.f, 0.f, 0.f};
  float* mb = &mbuf[wave * (2 * 16 * MSTR)];

  // Address selection instead of exec-masked loads: lanes 16..31 read a zero row.
  const _Float16* aBase  = (lane < 16) ? &lbuf[(sBase + lane) * LSTR] : zrow;
  const _Float16* b0Base = (lane < 16) ? &rbuf[(sBase + lane) * RSTR] : zrow;
  const _Float16* b1Base = (lane < 16) ? &rbuf[(sBase + 16 + lane) * RSTR] : zrow;

  // Loop-invariant LDS offsets (C-tile spill + band gather + gp_w rows).
  const int cstore = ((lane >> 4) * 8) * MSTR + (lane & 15);
  int bandOff[5];
  int gwBase[5];
#pragma unroll
  for (int j = 0; j < 5; ++j) {
    int d = dHalf + 2 * j;
    int rel = m - d - 1 + 16;  // in [6,30]
    bandOff[j] = (rel >> 4) * (16 * MSTR) + m * MSTR + (rel & 15);
    gwBase[j] = d * NHEAD;
  }

  for (int g = 0; g < NHEAD; ++g) {
    // A: 16x32 f16, head's 8 channels in K=0..7 (lanes 0..15 hold K 0..7,16..23)
    v8h av = *(const v8h*)(aBase + g * HSZ);
    v16h a;
#pragma unroll
    for (int j = 0; j < 8; ++j) {
      a[j] = av[j];
      a[j + 8] = (_Float16)0.0f;
    }
#pragma unroll
    for (int T = 0; T < 2; ++T) {
      // B: 32x16 f16, columns = rbuf cols [sBase+16T, +16) (lanes 0..15 hold K 0..7,16..23)
      v8h bv = *(const v8h*)((T ? b1Base : b0Base) + g * HSZ);
      v16h bm;
#pragma unroll
      for (int j = 0; j < 8; ++j) {
        bm[j] = bv[j];
        bm[j + 8] = (_Float16)0.0f;
      }
      v8f c = {};
      c = __builtin_amdgcn_wmma_f32_16x16x32_f16(false, a, false, bm,
                                                 (short)0, c, false, false);
      // spill C tile to LDS: element [M,N] -> lane = N + 16*(M>=8), vgpr = M&7
      float* mt = mb + T * (16 * MSTR) + cstore;
#pragma unroll
      for (int v = 0; v < 8; ++v) mt[v * MSTR] = c[v];
    }
    // band extraction: cost[m,d] = M_g[m, m-d-1]; fold gp_w[d,g]
#pragma unroll
    for (int j = 0; j < 5; ++j) acc[j] += gpwS[gwBase[j] + g] * mb[bandOff[j]];
  }

#pragma unroll
  for (int j = 0; j < 5; ++j) {
    int d = dHalf + 2 * j;
    Ob[(size_t)(CDIM + d) * planeHW + (w0 + sBase + m)] = acc[j] + gpbS[d];
  }
}

// ---------------- launch ----------------
extern "C" void kernel_launch(void* const* d_in, const int* in_sizes, int n_in,
                              void* d_out, int out_size, void* d_ws, size_t ws_size,
                              hipStream_t stream) {
  const float* left  = (const float*)d_in[0];
  const float* right = (const float*)d_in[1];
  const float* homo  = (const float*)d_in[2];
  const float* lnw   = (const float*)d_in[3];
  const float* lnb   = (const float*)d_in[4];
  const float* gpw   = (const float*)d_in[5];
  const float* gpb   = (const float*)d_in[6];
  float* out = (float*)d_out;
  float* ws  = (float*)d_ws;

  float* partL = ws;                 // 4*240*2 floats
  float* partR = ws + BATCH * NBLK * 2;
  float* stats = ws + 4096;          // 16 floats

  stats_partial_kernel<<<dim3(NBLK, BATCH), 256, 0, stream>>>(left, partL);
  stats_partial_kernel<<<dim3(NBLK, BATCH), 256, 0, stream>>>(right, partR);
  stats_reduce_kernel<<<8, 256, 0, stream>>>(ws, stats);
  fused_kernel<<<dim3(WDIM / TILE, HDIM, BATCH), 256, 0, stream>>>(
      left, right, homo, lnw, lnb, gpw, gpb, stats, out);
}
